// Glm4MoeGroupedExperts_35691178230102
// MI455X (gfx1250) — matrix operand order repaired and checked by reading, not compiled, not run on hardware
//
#include <hip/hip_runtime.h>
#include <hip/hip_bf16.h>

// ---------------- problem constants (from reference) ----------------
#define EXP   16
#define HDIM  2048
#define IDIM  1408
#define TOPK  4
#define NTOK  2048          // B*S
#define NP    8192          // NTOK*TOPK dispatched rows

// ---------------- tiling ----------------
#define TM 128
#define TN 64
#define TK 32
#define MAXTILES 80         // worst case: NP/TM + EXP = 64+16

// ---------------- workspace layout ----------------
// int region (int32 indices):
#define WS_CNT     0
#define WS_FILL    16
#define WS_OFF     32
#define WS_NT      64
#define WS_TEXP    128      // [96]
#define WS_TROW    224      // [96]
#define WS_TCNT    320      // [96]
#define WS_ROWTOK  1024     // [8192]
#define WS_ROWW    9216     // [8192]  (floats)
// bf16 region (ushort indices from ws base):
#define US_XBF   34816u                            // NTOK*HDIM     (8 MB)
#define US_HACT  (US_XBF  + (unsigned)NTOK*HDIM)   // NP*IDIM       (23 MB)
#define US_GWB   (US_HACT + (unsigned)NP*IDIM)     // EXP*IDIM*HDIM (92 MB)
#define US_UWB   (US_GWB  + (unsigned)EXP*IDIM*HDIM)
#define US_DWB   (US_UWB  + (unsigned)EXP*IDIM*HDIM)
// total ws requirement ~ 309 MB

typedef __attribute__((ext_vector_type(16))) __bf16        v16bf;
typedef __attribute__((ext_vector_type(8)))  float         v8f;
typedef __attribute__((ext_vector_type(4)))  unsigned int  v4u;
typedef __attribute__((ext_vector_type(8)))  int           v8i;
typedef __attribute__((ext_vector_type(4)))  int           v4i;

#define WMMA_BF16(a, b, c) \
  __builtin_amdgcn_wmma_f32_16x16x32_bf16(false, (a), false, (b), (short)0, (c), false, false)

// ---------------- TDM support (guarded; falls back to copy staging) ----------------
#if defined(__has_builtin)
#  if __has_builtin(__builtin_amdgcn_tensor_load_to_lds)
#    define HAVE_TDM 1
#  endif
#endif

#ifdef HAVE_TDM
__device__ __forceinline__ void tdm_issue(v4u g0, v8i g1) {
  v4i gz = {0, 0, 0, 0};
#if __has_include(<hip/amd_detail/amd_gfx1250_TDM.h>)
  // amdgpu-toolchain (clang-23): 6-arg form
  v8i gz8 = {0, 0, 0, 0, 0, 0, 0, 0};
  __builtin_amdgcn_tensor_load_to_lds(g0, g1, gz, gz, gz8, 0);
#else
  // ROCm 7.2 (clang-22): 5-arg form
  __builtin_amdgcn_tensor_load_to_lds(g0, g1, gz, gz, 0);
#endif
}

// 2-D tile load global->LDS: tile_d0 x tile_d1 bf16 elements, row stride stride0.
// D# packing per CDNA5 ISA 8.3/8.4 (data_size=1 -> 2 bytes).
__device__ __forceinline__ void tdm_load_2d(unsigned lds_addr, const void* gptr,
                                            unsigned tensor_d0, unsigned tensor_d1,
                                            unsigned tile_d0, unsigned tile_d1,
                                            unsigned stride0) {
  unsigned long long ga = (unsigned long long)(uintptr_t)gptr;
  v4u g0;
  g0[0] = 1u;                                            // count=1, user descriptor
  g0[1] = lds_addr;                                      // LDS byte address
  g0[2] = (unsigned)(ga & 0xFFFFFFFFu);                  // global_addr[31:0]
  g0[3] = (unsigned)((ga >> 32) & 0x1FFFFFFu) | 0x80000000u; // addr[56:32] | type=2
  v8i g1;
  g1[0] = 0x10000;                                       // data_size=1 (2B)
  g1[1] = (int)((tensor_d0 & 0xFFFFu) << 16);            // tensor_dim0[15:0]
  g1[2] = (int)(((tensor_d0 >> 16) & 0xFFFFu) | ((tensor_d1 & 0xFFFFu) << 16));
  g1[3] = (int)(((tensor_d1 >> 16) & 0xFFFFu) | (tile_d0 << 16)); // tile_dim0
  g1[4] = (int)tile_d1;                                  // tile_dim1 (tile_dim2=0)
  g1[5] = (int)stride0;                                  // tensor_dim0_stride[31:0]
  g1[6] = 0;
  g1[7] = 0;
  tdm_issue(g0, g1);
}
#endif  // HAVE_TDM

// fp32 -> bf16 round-to-nearest-even
__device__ __forceinline__ unsigned short f2bf(float f) {
  unsigned u = __float_as_uint(f);
  unsigned r = ((u >> 16) & 1u) + 0x7fffu;
  return (unsigned short)((u + r) >> 16);
}

__device__ __forceinline__ void cvt8(unsigned short* d, const float* s) {
  float4 a = *(const float4*)s;
  float4 b = *(const float4*)(s + 4);
  union { unsigned short h[8]; uint4 q; } t;
  t.h[0] = f2bf(a.x); t.h[1] = f2bf(a.y); t.h[2] = f2bf(a.z); t.h[3] = f2bf(a.w);
  t.h[4] = f2bf(b.x); t.h[5] = f2bf(b.y); t.h[6] = f2bf(b.z); t.h[7] = f2bf(b.w);
  *(uint4*)d = t.q;
}

__device__ __forceinline__ void cp8(unsigned short* d, const unsigned short* s) {
  *(uint4*)d = *(const uint4*)s;
}
__device__ __forceinline__ void cp16(unsigned short* d, const unsigned short* s) {
  ((uint4*)d)[0] = ((const uint4*)s)[0];
  ((uint4*)d)[1] = ((const uint4*)s)[1];
}

__device__ __forceinline__ float fast_silu(float g) {
  return g * __builtin_amdgcn_rcpf(1.f + __expf(-g));
}

union Frag { uint4 q[2]; v16bf v; };

__device__ __forceinline__ v16bf load_frag_a(const unsigned short* sA, int rowBase, int lane) {
  int m = lane & 15, hi = lane >> 4;
  const unsigned short* p = sA + (rowBase + m) * TK + hi * 8;
  Frag f;
  f.q[0] = *(const uint4*)(p);
  f.q[1] = *(const uint4*)(p + 16);
  return f.v;
}

__device__ __forceinline__ v16bf load_frag_b(const unsigned short* sB, int nBase, int lane) {
  int n = lane & 15, kh = lane >> 4;
  const unsigned short* p = sB + (nBase + n) * TK + kh * 8;
  Frag f;
  f.q[0] = *(const uint4*)(p);
  f.q[1] = *(const uint4*)(p + 16);
  return f.v;
}

// ---------------- one-time fp32 -> bf16 conversion ----------------
__global__ __launch_bounds__(256) void k_cvt(const float* __restrict__ s,
                                             unsigned short* __restrict__ d) {
  size_t i = ((size_t)blockIdx.x * blockDim.x + threadIdx.x) * 8;
  cvt8(d + i, s + i);
}

// ---------------- routing kernels ----------------
__global__ void k_zero_counters(int* ws) {
  int t = threadIdx.x;
  if (t < 32) ws[WS_CNT + t] = 0;
}

__global__ void k_count(const int* __restrict__ idx, int* __restrict__ ws) {
  int p = blockIdx.x * blockDim.x + threadIdx.x;
  if (p < NP) atomicAdd(&ws[WS_CNT + idx[p]], 1);
}

__global__ void k_prefix(int* ws) {
  int acc = 0, nt = 0;
  ws[WS_OFF] = 0;
  for (int e = 0; e < EXP; ++e) {
    int c = ws[WS_CNT + e];
    for (int r = 0; r < c; r += TM) {
      ws[WS_TEXP + nt] = e;
      ws[WS_TROW + nt] = acc + r;
      ws[WS_TCNT + nt] = (c - r < TM) ? (c - r) : TM;
      ++nt;
    }
    acc += c;
    ws[WS_OFF + e + 1] = acc;
  }
  ws[WS_NT] = nt;
}

__global__ void k_scatter(const int* __restrict__ idx, const float* __restrict__ tw,
                          int* __restrict__ ws) {
  int p = blockIdx.x * blockDim.x + threadIdx.x;
  if (p < NP) {
    int e = idx[p];
    int pos = atomicAdd(&ws[WS_FILL + e], 1);
    int row = ws[WS_OFF + e] + pos;
    ws[WS_ROWTOK + row] = p >> 2;               // TOPK = 4
    ((float*)ws)[WS_ROWW + row] = tw[p];
  }
}

__global__ void k_zero_out(float4* __restrict__ out) {
  size_t i = (size_t)blockIdx.x * blockDim.x + threadIdx.x;
  out[i] = make_float4(0.f, 0.f, 0.f, 0.f);
}

// ---------------- grouped GEMM 1: gate & up + SiLU ----------------
__global__ __launch_bounds__(256) void moe_gateup(
    const unsigned short* __restrict__ xb, const unsigned short* __restrict__ gwb,
    const unsigned short* __restrict__ uwb, const int* __restrict__ ws,
    unsigned short* __restrict__ hact) {
  int tile = blockIdx.y;
  if (tile >= ws[WS_NT]) return;
  int e         = ws[WS_TEXP + tile];
  int rowStart  = ws[WS_TROW + tile];
  int rowsValid = ws[WS_TCNT + tile];
  int nBase     = blockIdx.x * TN;              // column in IDIM

  __shared__ unsigned short sA [2][TM * TK];    // 2 x 8 KB
  __shared__ unsigned short sBg[2][TN * TK];    // 2 x 4 KB
  __shared__ unsigned short sBu[2][TN * TK];    // 2 x 4 KB

  int tid = threadIdx.x;
  int lane = tid & 31, wid = tid >> 5;
  int waveM = wid & 3, waveN = wid >> 2;

  // A staging (token-gathered rows -> manual copy): row tid>>1, 16 bf16 at (tid&1)*16
  int aRowIdx = tid >> 1;
  int akq     = (tid & 1) * 16;
  int aR      = (aRowIdx < rowsValid) ? aRowIdx : 0;
  int tok     = ws[WS_ROWTOK + rowStart + aR];
  const unsigned short* xrow = xb + (size_t)tok * HDIM + akq;
  int aOff = aRowIdx * TK + akq;

  // B tiles: dense 64x32 bf16 slabs of the expert weights
  const unsigned short* gtile = gwb + ((size_t)e * IDIM + (size_t)nBase) * HDIM;
  const unsigned short* utile = uwb + ((size_t)e * IDIM + (size_t)nBase) * HDIM;
#ifndef HAVE_TDM
  int bRow = tid >> 2;
  int bkq  = (tid & 3) * 8;
  int bOff = bRow * TK + bkq;
  const unsigned short* grow = gtile + (size_t)bRow * HDIM + bkq;
  const unsigned short* urow = utile + (size_t)bRow * HDIM + bkq;
#endif

  auto stage = [&](int buf, int k0) {
    cp16(&sA[buf][aOff], xrow + k0);
    __builtin_prefetch(xrow + k0 + 2 * TK, 0, 0);
#ifdef HAVE_TDM
    if (wid == 0) {   // one wave issues the DMA; EXEC ignored by TDM
      tdm_load_2d((unsigned)(uintptr_t)&sBg[buf][0], gtile + k0,
                  HDIM, IDIM, TK, TN, HDIM);
      tdm_load_2d((unsigned)(uintptr_t)&sBu[buf][0], utile + k0,
                  HDIM, IDIM, TK, TN, HDIM);
    }
#else
    cp8(&sBg[buf][bOff], grow + k0);
    cp8(&sBu[buf][bOff], urow + k0);
    __builtin_prefetch(grow + k0 + 2 * TK, 0, 0);
    __builtin_prefetch(urow + k0 + 2 * TK, 0, 0);
#endif
  };

  v8f ag[2][2] = {{}, {}};
  v8f au[2][2] = {{}, {}};

  const int NK = HDIM / TK;
  stage(0, 0);
  for (int ki = 0; ki < NK; ++ki) {
#ifdef HAVE_TDM
    if (wid == 0) __builtin_amdgcn_s_wait_tensorcnt(0);
#endif
    __syncthreads();
    int cur = ki & 1;
    if (ki + 1 < NK) stage(cur ^ 1, (ki + 1) * TK);

    const unsigned short* Ab = sA[cur];
    v16bf bg0 = load_frag_b(sBg[cur], waveN * 32,      lane);
    v16bf bg1 = load_frag_b(sBg[cur], waveN * 32 + 16, lane);
    v16bf bu0 = load_frag_b(sBu[cur], waveN * 32,      lane);
    v16bf bu1 = load_frag_b(sBu[cur], waveN * 32 + 16, lane);
#pragma unroll
    for (int mt = 0; mt < 2; ++mt) {
      v16bf a = load_frag_a(Ab, waveM * 32 + mt * 16, lane);
      ag[mt][0] = WMMA_BF16(a, bg0, ag[mt][0]);
      ag[mt][1] = WMMA_BF16(a, bg1, ag[mt][1]);
      au[mt][0] = WMMA_BF16(a, bu0, au[mt][0]);
      au[mt][1] = WMMA_BF16(a, bu1, au[mt][1]);
    }
  }

  // epilogue: h = silu(gate) * up, store bf16
  int hi = lane >> 4, n = lane & 15;
#pragma unroll
  for (int mt = 0; mt < 2; ++mt) {
#pragma unroll
    for (int nt = 0; nt < 2; ++nt) {
      int colBase = nBase + waveN * 32 + nt * 16 + n;
#pragma unroll
      for (int r = 0; r < 8; ++r) {
        int mrow = waveM * 32 + mt * 16 + 8 * hi + r;
        if (mrow < rowsValid) {
          size_t base = (size_t)(rowStart + mrow) * IDIM;
          hact[base + colBase] = f2bf(fast_silu(ag[mt][nt][r]) * au[mt][nt][r]);
        }
      }
    }
  }
}

// ---------------- grouped GEMM 2: down proj + weighted scatter-add ----------------
__global__ __launch_bounds__(256) void moe_down(
    const unsigned short* __restrict__ hact, const unsigned short* __restrict__ dwb,
    const int* __restrict__ ws, float* __restrict__ out) {
  int tile = blockIdx.y;
  if (tile >= ws[WS_NT]) return;
  int e         = ws[WS_TEXP + tile];
  int rowStart  = ws[WS_TROW + tile];
  int rowsValid = ws[WS_TCNT + tile];
  int nBase     = blockIdx.x * TN;              // column in HDIM

  __shared__ unsigned short sA[2][TM * TK];     // 2 x 8 KB
  __shared__ unsigned short sB[2][TN * TK];     // 2 x 4 KB

  int tid = threadIdx.x;
  int lane = tid & 31, wid = tid >> 5;
  int waveM = wid & 3, waveN = wid >> 2;

  // A: contiguous sorted rows of hact -> dense 128x32 tile
  const unsigned short* atile = hact + (size_t)rowStart * IDIM;
  const unsigned short* btile = dwb + ((size_t)e * HDIM + (size_t)nBase) * IDIM;
#ifndef HAVE_TDM
  int aRowIdx = tid >> 1;
  int akq     = (tid & 1) * 16;
  int aR      = (aRowIdx < rowsValid) ? aRowIdx : 0;
  const unsigned short* hrow = atile + (size_t)aR * IDIM + akq;
  int aOff = aRowIdx * TK + akq;
  int bRow = tid >> 2;
  int bkq  = (tid & 3) * 8;
  const unsigned short* wrow = btile + (size_t)bRow * IDIM + bkq;
  int bOff = bRow * TK + bkq;
#endif

  auto stage = [&](int buf, int k0) {
#ifdef HAVE_TDM
    if (wid == 0) {
      tdm_load_2d((unsigned)(uintptr_t)&sA[buf][0], atile + k0,
                  IDIM, NP,   TK, TM, IDIM);
      tdm_load_2d((unsigned)(uintptr_t)&sB[buf][0], btile + k0,
                  IDIM, HDIM, TK, TN, IDIM);
    }
#else
    cp16(&sA[buf][aOff], hrow + k0);
    cp8 (&sB[buf][bOff], wrow + k0);
    __builtin_prefetch(hrow + k0 + 2 * TK, 0, 0);
    __builtin_prefetch(wrow + k0 + 2 * TK, 0, 0);
#endif
  };

  v8f acc[2][2] = {{}, {}};

  const int NK = IDIM / TK;
  stage(0, 0);
  for (int ki = 0; ki < NK; ++ki) {
#ifdef HAVE_TDM
    if (wid == 0) __builtin_amdgcn_s_wait_tensorcnt(0);
#endif
    __syncthreads();
    int cur = ki & 1;
    if (ki + 1 < NK) stage(cur ^ 1, (ki + 1) * TK);

    const unsigned short* Ab = sA[cur];
    v16bf b0 = load_frag_b(sB[cur], waveN * 32,      lane);
    v16bf b1 = load_frag_b(sB[cur], waveN * 32 + 16, lane);
#pragma unroll
    for (int mt = 0; mt < 2; ++mt) {
      v16bf a = load_frag_a(Ab, waveM * 32 + mt * 16, lane);
      acc[mt][0] = WMMA_BF16(a, b0, acc[mt][0]);
      acc[mt][1] = WMMA_BF16(a, b1, acc[mt][1]);
    }
  }

  int hi = lane >> 4, n = lane & 15;
#pragma unroll
  for (int mt = 0; mt < 2; ++mt) {
#pragma unroll
    for (int nt = 0; nt < 2; ++nt) {
      int colBase = nBase + waveN * 32 + nt * 16 + n;
#pragma unroll
      for (int r = 0; r < 8; ++r) {
        int mrow = waveM * 32 + mt * 16 + 8 * hi + r;
        if (mrow < rowsValid) {
          int grow = rowStart + mrow;
          int tok  = ws[WS_ROWTOK + grow];
          float w  = ((const float*)ws)[WS_ROWW + grow];
          atomicAdd(out + (size_t)tok * HDIM + colBase, acc[mt][nt][r] * w);
        }
      }
    }
  }
}

// ---------------- host launch ----------------
extern "C" void kernel_launch(void* const* d_in, const int* in_sizes, int n_in,
                              void* d_out, int out_size, void* d_ws, size_t ws_size,
                              hipStream_t stream) {
  const float* x    = (const float*)d_in[0];
  const int*   tidx = (const int*)  d_in[1];
  const float* tw   = (const float*)d_in[2];
  const float* gw   = (const float*)d_in[3];
  const float* uw   = (const float*)d_in[4];
  const float* dw   = (const float*)d_in[5];
  float* out = (float*)d_out;
  int* wsi = (int*)d_ws;
  unsigned short* usw = (unsigned short*)d_ws;
  unsigned short* xb   = usw + US_XBF;
  unsigned short* hact = usw + US_HACT;
  unsigned short* gwb  = usw + US_GWB;
  unsigned short* uwb  = usw + US_UWB;
  unsigned short* dwb  = usw + US_DWB;

  // routing
  k_zero_counters<<<1, 64, 0, stream>>>(wsi);
  k_count  <<<NP / 256, 256, 0, stream>>>(tidx, wsi);
  k_prefix <<<1, 1, 0, stream>>>(wsi);
  k_scatter<<<NP / 256, 256, 0, stream>>>(tidx, tw, wsi);
  k_zero_out<<<(NTOK * HDIM) / (4 * 256), 256, 0, stream>>>((float4*)out);

  // one-time fp32 -> bf16 conversion (2048 fp32 per block)
  k_cvt<<<(NTOK * HDIM) / 2048, 256, 0, stream>>>(x, xb);
  k_cvt<<<(EXP * IDIM * HDIM) / 2048, 256, 0, stream>>>(gw, gwb);
  k_cvt<<<(EXP * IDIM * HDIM) / 2048, 256, 0, stream>>>(uw, uwb);
  k_cvt<<<(EXP * HDIM * IDIM) / 2048, 256, 0, stream>>>(dw, dwb);

  dim3 g1(IDIM / TN, MAXTILES);
  moe_gateup<<<g1, 256, 0, stream>>>(xb, gwb, uwb, wsi, hact);

  dim3 g2(HDIM / TN, MAXTILES);
  moe_down<<<g2, 256, 0, stream>>>(hact, dwb, wsi, out);
}
